// HeteroRGCNLayer_38663295599335
// MI455X (gfx1250) — matrix operand order repaired and checked by reading, not compiled, not run on hardware
//
#include <hip/hip_runtime.h>

// Problem constants (from reference)
#define NT 20000
#define NC 80000
#define DD 256
#define SLOPE 0.01f

typedef __attribute__((ext_vector_type(16))) _Float16 v16h;
typedef __attribute__((ext_vector_type(8)))  _Float16 v8h;
typedef __attribute__((ext_vector_type(8)))  float    v8f;

// ---------------- elementwise helpers ----------------

__global__ __launch_bounds__(256) void f32_to_f16_kernel(const float* __restrict__ in,
                                                         _Float16* __restrict__ out, int n) {
  int i = blockIdx.x * 256 + threadIdx.x;
  if (i < n) out[i] = (_Float16)in[i];
}

__global__ __launch_bounds__(256) void zero_f32_kernel(float* __restrict__ p, long n) {
  long i = (long)blockIdx.x * 256 + threadIdx.x;
  if (i < n) p[i] = 0.0f;
}

// ---------------- WMMA GEMM: Y = lrelu(X @ W^T + b) [+ residual] ----------------
// X: [N, 256] f16 row-major, W: [256, 256] f16 row-major (we compute X @ W^T, so
// C[i,j] = sum_k X[i,k] * W[j,k] — both operands contiguous along K).
// One wave computes one 16x16 output tile; block = 128 threads = 4 waves covering
// 16 rows x 64 cols; grid = (N/16, 4).
template <bool WRITE_F16>
__global__ __launch_bounds__(128) void gemm_lrelu_kernel(
    const _Float16* __restrict__ X, const _Float16* __restrict__ W,
    const float* __restrict__ bias,
    _Float16* __restrict__ Y16, float* __restrict__ Y32,
    const float* __restrict__ residual) {
  const int wave     = threadIdx.x >> 5;
  const int lane     = threadIdx.x & 31;
  const int half     = lane >> 4;          // 0: lanes 0-15, 1: lanes 16-31
  const int lr       = lane & 15;
  const int row_base = blockIdx.x * 16;
  const int col_base = (blockIdx.y * 4 + wave) * 16;

  const _Float16* __restrict__ arow = X + (size_t)(row_base + lr) * DD;
  const _Float16* __restrict__ brow = W + (size_t)(col_base + lr) * DD;

  v8f acc = {};
#pragma unroll
  for (int k0 = 0; k0 < DD; k0 += 32) {
    // A fragment (16-bit A 16x32 layout): lanes 0-15 hold K=[k0+0..7] in v0-3 and
    // K=[k0+16..23] in v4-7; lanes 16-31 hold K=[k0+8..15] and K=[k0+24..31].
    v8h a_lo = *(const v8h*)(arow + k0 + half * 8);
    v8h a_hi = *(const v8h*)(arow + k0 + half * 8 + 16);
    // B fragment (32x16): lane j = column j; same K order as A: lanes 0-15 hold
    // K=[k0+0..7] and K=[k0+16..23]; lanes 16-31 hold K=[k0+8..15] and K=[k0+24..31].
    v8h b_lo = *(const v8h*)(brow + k0 + half * 8);
    v8h b_hi = *(const v8h*)(brow + k0 + half * 8 + 16);

    v16h a = __builtin_shufflevector(a_lo, a_hi, 0,1,2,3,4,5,6,7,8,9,10,11,12,13,14,15);
    v16h b = __builtin_shufflevector(b_lo, b_hi, 0,1,2,3,4,5,6,7,8,9,10,11,12,13,14,15);

    acc = __builtin_amdgcn_wmma_f32_16x16x32_f16(
        /*neg_a=*/false, a, /*neg_b=*/false, b,
        /*c_mod=*/(short)0, acc, /*reuse_a=*/false, /*reuse_b=*/false);
  }

  // C/D layout: VGPR r, lanes 0-15 -> (M=r, N=lane); lanes 16-31 -> (M=8+r, N=lane-16)
  const int col_o = col_base + lr;
  const float bcol = bias[col_o];
#pragma unroll
  for (int r = 0; r < 8; ++r) {
    const int row_o = row_base + r + half * 8;
    float v = acc[r] + bcol;
    v = (v >= 0.0f) ? v : SLOPE * v;
    const size_t idx = (size_t)row_o * DD + col_o;
    if (WRITE_F16) {
      Y16[idx] = (_Float16)v;
    } else {
      Y32[idx] = v + residual[idx];
    }
  }
}

// ---------------- edge scatter: segment-sum of messages + degree count ----------------
__global__ __launch_bounds__(256) void scatter_kernel(
    const _Float16* __restrict__ Wh, const int* __restrict__ src,
    const int* __restrict__ dst, float* __restrict__ acc,
    float* __restrict__ cnt, int nE) {
  const int e = blockIdx.x;
  if (e >= nE) return;
  const int s = src[e];
  const int d = dst[e];
  const int j = threadIdx.x;
  const float v = (float)Wh[(size_t)s * DD + j];
  atomicAdd(acc + (size_t)d * DD + j, v);
  if (j == 0) atomicAdd(cnt + d, 1.0f);
}

// ---------------- cross-etype mean combine, emit f16 h ----------------
__global__ __launch_bounds__(256) void combine_kernel(
    const float* __restrict__ accA, const float* __restrict__ cntA,
    const float* __restrict__ accB, const float* __restrict__ cntB,
    _Float16* __restrict__ h16, int n_nodes) {
  long i = (long)blockIdx.x * 256 + threadIdx.x;
  if (i >= (long)n_nodes * DD) return;
  const int node = (int)(i / DD);
  const float a = accA[i] / fmaxf(cntA[node], 1.0f);
  const float b = accB[i] / fmaxf(cntB[node], 1.0f);
  h16[i] = (_Float16)(0.5f * (a + b));
}

// ---------------- host orchestration ----------------
extern "C" void kernel_launch(void* const* d_in, const int* in_sizes, int n_in,
                              void* d_out, int out_size, void* d_ws, size_t ws_size,
                              hipStream_t stream) {
  const float* feat_t = (const float*)d_in[0];
  const float* feat_c = (const float*)d_in[1];
  const float* W_t2c  = (const float*)d_in[2];
  const float* b_t2c  = (const float*)d_in[3];
  const float* W_c2t  = (const float*)d_in[4];
  const float* b_c2t  = (const float*)d_in[5];
  const float* W_c2c  = (const float*)d_in[6];
  const float* b_c2c  = (const float*)d_in[7];
  const float* W_t2t  = (const float*)d_in[8];
  const float* b_t2t  = (const float*)d_in[9];
  const float* W_h    = (const float*)d_in[10];
  const float* b_h    = (const float*)d_in[11];
  const int* src_t2c  = (const int*)d_in[12];
  const int* dst_t2c  = (const int*)d_in[13];
  const int* src_c2t  = (const int*)d_in[14];
  const int* dst_c2t  = (const int*)d_in[15];
  const int* src_c2c  = (const int*)d_in[16];
  const int* dst_c2c  = (const int*)d_in[17];
  const int* src_t2t  = (const int*)d_in[18];
  const int* dst_t2t  = (const int*)d_in[19];

  const int E_t2c = in_sizes[12];   // 320000
  const int E_c2t = in_sizes[14];   // 320000
  const int E_c2c = in_sizes[16];   // 640000
  const int E_t2t = in_sizes[18];   // 160000

  // ---- carve workspace ----
  char* p = (char*)d_ws;
  auto carve = [&](size_t bytes) -> void* {
    void* r = (void*)p;
    p += (bytes + 255) & ~(size_t)255;
    return r;
  };
  _Float16* x16_t   = (_Float16*)carve((size_t)NT * DD * 2);
  _Float16* x16_c   = (_Float16*)carve((size_t)NC * DD * 2);
  _Float16* w16_t2c = (_Float16*)carve((size_t)DD * DD * 2);
  _Float16* w16_c2t = (_Float16*)carve((size_t)DD * DD * 2);
  _Float16* w16_c2c = (_Float16*)carve((size_t)DD * DD * 2);
  _Float16* w16_t2t = (_Float16*)carve((size_t)DD * DD * 2);
  _Float16* w16_h   = (_Float16*)carve((size_t)DD * DD * 2);
  _Float16* wh_t2c  = (_Float16*)carve((size_t)NT * DD * 2);
  _Float16* wh_c2t  = (_Float16*)carve((size_t)NC * DD * 2);
  _Float16* wh_c2c  = (_Float16*)carve((size_t)NC * DD * 2);
  _Float16* wh_t2t  = (_Float16*)carve((size_t)NT * DD * 2);
  // accumulators + counters carved contiguously so one zero-kernel clears all
  float* zero_base  = (float*)p;
  float* acc_c2t = (float*)carve((size_t)NT * DD * 4);
  float* acc_t2t = (float*)carve((size_t)NT * DD * 4);
  float* acc_t2c = (float*)carve((size_t)NC * DD * 4);
  float* acc_c2c = (float*)carve((size_t)NC * DD * 4);
  float* cnt_c2t = (float*)carve((size_t)NT * 4);
  float* cnt_t2t = (float*)carve((size_t)NT * 4);
  float* cnt_t2c = (float*)carve((size_t)NC * 4);
  float* cnt_c2c = (float*)carve((size_t)NC * 4);
  const long zero_floats = (long)(((char*)p - (char*)zero_base) / 4);

  _Float16* h16_t = (_Float16*)carve((size_t)NT * DD * 2);
  _Float16* h16_c = (_Float16*)carve((size_t)NC * DD * 2);

  // ---- 1) f32 -> f16 conversions ----
  auto cvt = [&](const float* in, _Float16* out, int n) {
    f32_to_f16_kernel<<<(n + 255) / 256, 256, 0, stream>>>(in, out, n);
  };
  cvt(feat_t, x16_t, NT * DD);
  cvt(feat_c, x16_c, NC * DD);
  cvt(W_t2c, w16_t2c, DD * DD);
  cvt(W_c2t, w16_c2t, DD * DD);
  cvt(W_c2c, w16_c2c, DD * DD);
  cvt(W_t2t, w16_t2t, DD * DD);
  cvt(W_h,   w16_h,   DD * DD);

  // ---- 2) zero accumulators/counters (must happen every call: graph replays) ----
  zero_f32_kernel<<<(int)((zero_floats + 255) / 256), 256, 0, stream>>>(zero_base, zero_floats);

  // ---- 3) per-etype Linear + leaky_relu via WMMA ----
  dim3 gT(NT / 16, 4), gC(NC / 16, 4);
  gemm_lrelu_kernel<true><<<gT, 128, 0, stream>>>(x16_t, w16_t2c, b_t2c, wh_t2c, nullptr, nullptr);
  gemm_lrelu_kernel<true><<<gC, 128, 0, stream>>>(x16_c, w16_c2t, b_c2t, wh_c2t, nullptr, nullptr);
  gemm_lrelu_kernel<true><<<gC, 128, 0, stream>>>(x16_c, w16_c2c, b_c2c, wh_c2c, nullptr, nullptr);
  gemm_lrelu_kernel<true><<<gT, 128, 0, stream>>>(x16_t, w16_t2t, b_t2t, wh_t2t, nullptr, nullptr);

  // ---- 4) edge scatter (segment sums + degrees) ----
  scatter_kernel<<<E_t2c, 256, 0, stream>>>(wh_t2c, src_t2c, dst_t2c, acc_t2c, cnt_t2c, E_t2c);
  scatter_kernel<<<E_c2t, 256, 0, stream>>>(wh_c2t, src_c2t, dst_c2t, acc_c2t, cnt_c2t, E_c2t);
  scatter_kernel<<<E_c2c, 256, 0, stream>>>(wh_c2c, src_c2c, dst_c2c, acc_c2c, cnt_c2c, E_c2c);
  scatter_kernel<<<E_t2t, 256, 0, stream>>>(wh_t2t, src_t2t, dst_t2t, acc_t2t, cnt_t2t, E_t2t);

  // ---- 5) per-etype mean + cross-etype mean ----
  combine_kernel<<<(NT * DD + 255) / 256, 256, 0, stream>>>(acc_c2t, cnt_c2t, acc_t2t, cnt_t2t, h16_t, NT);
  combine_kernel<<<(NC * DD + 255) / 256, 256, 0, stream>>>(acc_t2c, cnt_t2c, acc_c2c, cnt_c2c, h16_c, NC);

  // ---- 6) out = lrelu(h @ W_h^T + b_h) + feat ----
  float* out_t = (float*)d_out;
  float* out_c = out_t + (size_t)NT * DD;
  gemm_lrelu_kernel<false><<<gT, 128, 0, stream>>>(h16_t, w16_h, b_h, nullptr, out_t, feat_t);
  gemm_lrelu_kernel<false><<<gC, 128, 0, stream>>>(h16_c, w16_h, b_h, nullptr, out_c, feat_c);
}
